// ExecutionEngine_76716705841556
// MI455X (gfx1250) — matrix operand order repaired and checked by reading, not compiled, not run on hardware
//
#include <hip/hip_runtime.h>
#include <hip/hip_bf16.h>

// ---------------------------------------------------------------------------
// bf16 WMMA implementation of the routed ExecutionEngine pipeline.
// Weights pre-packed into WMMA A-fragment order; activations kept channel-
// contiguous [b][p][c]; conv input tiles staged into LDS with the CDNA5
// async global->LDS data mover (ASYNCcnt) instead of VGPR round-trips.
// ---------------------------------------------------------------------------

typedef __attribute__((ext_vector_type(16))) __bf16 v16bf;
typedef __attribute__((ext_vector_type(8)))  float  v8f;

union BFrag {
    v16bf v;
    unsigned int u32[8];
    unsigned short u16[16];
    uint4 q[2];
};

__device__ inline unsigned short f2bf_bits(float f) {
    union { float f; unsigned u; } a; a.f = f;
    unsigned r = a.u + 0x7FFFu + ((a.u >> 16) & 1u);   // round-to-nearest-even
    return (unsigned short)(r >> 16);
}
__device__ inline float bfbits2f(unsigned short h) {
    union { float f; unsigned u; } a; a.u = ((unsigned)h) << 16; return a.f;
}

// CDNA5 async global->LDS copy (16B per enabled lane), tracked by ASYNCcnt.
__device__ inline void async_copy_b128(unsigned lds_off, unsigned long long gaddr) {
    asm volatile("global_load_async_to_lds_b128 %0, %1, off"
                 :: "v"(lds_off), "v"(gaddr)
                 : "memory");
}
__device__ inline void wait_async0() {
    asm volatile("s_wait_asynccnt 0" ::: "memory");
}

#define CHUNK 128          // Cin channels staged in LDS per chunk
#define NT    13           // 196 pixels -> 13 tiles of 16
#define HW    196
#define LSEQ  8            // program length L

// ---------------------------------------------------------------------------
// Pack f32 weights [rows=E*Cout][Cin][KKK] into bf16 WMMA A-fragments.
// Fragment order: ((rowblk*KKK + tap)*(Cin/32) + ks) ; 32 lanes x 16 bf16.
// ---------------------------------------------------------------------------
__global__ void pack_w_kernel(const float* __restrict__ W,
                              unsigned short* __restrict__ Wp,
                              int Cin, int KKK, int total)
{
    const int i = blockIdx.x * blockDim.x + threadIdx.x;
    if (i >= total) return;
    const int kmapA[8] = {0, 2, 4, 6, 16, 18, 20, 22};
    const int lane = i & 31;
    int rest = i >> 5;
    const int nKs = Cin >> 5;
    const int ks  = rest % nKs; rest /= nKs;
    const int tap = rest % KKK; rest /= KKK;
    const int half = lane >> 4, l15 = lane & 15;
    const int row  = rest * 16 + l15;          // rest = row-block over E*Cout
    const float* wr = W + (size_t)row * Cin * KKK + tap;
    BFrag f;
#pragma unroll
    for (int v = 0; v < 8; ++v) {
        const int ci = ks * 32 + kmapA[v] + 8 * half;
        f.u16[2 * v]     = f2bf_bits(wr[(size_t)ci * KKK]);
        f.u16[2 * v + 1] = f2bf_bits(wr[(size_t)(ci + 1) * KKK]);
    }
    uint4* dst = (uint4*)(Wp + (size_t)i * 16);
    dst[0] = f.q[0];
    dst[1] = f.q[1];
}

// ---------------------------------------------------------------------------
// Routed conv layer as shifted-tap WMMA GEMM.
//   grid.x = batch, grid.y = Cout/128, block = 256 (8 waves)
//   Activations: bf16 [b][p=196][c] (channel-contiguous). Input staged in LDS
//   as zero-haloed [16][16][CHUNK] bf16 so taps are plain offsets.
// ---------------------------------------------------------------------------
__global__ __launch_bounds__(256)
void conv_wmma(const float* __restrict__ Xf,            // f32 img [b][c][p] or null
               const unsigned short* __restrict__ Xh,   // bf16 [b][p][c] or null
               const unsigned short* __restrict__ Wpk,  // packed A-fragments
               const float* __restrict__ Bias,          // [E][Cout]
               const unsigned short* __restrict__ Res,  // bf16 [b][p][Cout] or null
               unsigned short* __restrict__ Out,        // bf16 [b][p][Cout]
               const int* __restrict__ pinds,
               int step, int sub,                       // widx = pinds[b*L+step]-sub
               int CinW, int CinData, int cinMod,       // weight-Cin, data-Cin, concat mod
               int Cout, int KK)
{
    const int b    = blockIdx.x;
    const int tid  = threadIdx.x;
    const int lane = tid & 31;
    const int wave = tid >> 5;
    const int half = lane >> 4;
    const int l15  = lane & 15;
    const int KKK  = KK * KK;
    const int nCk  = CinW / CHUNK;
    const int nKsTot = CinW >> 5;
    const int nCo  = Cout >> 4;
    const int widx = (step >= 0) ? (pinds[b * LSEQ + step] - sub) : 0;
    const int cb   = blockIdx.y * 8 + wave;        // 16-row Cout block index

    __shared__ unsigned short lds[256 * CHUNK];    // 64 KB padded 16x16 x 128ch

    const float* Bp = Bias + (size_t)widx * Cout;
    const int rowb  = cb * 16 + 8 * half;          // first of this lane-half's 8 rows

    v8f acc[NT];
#pragma unroll
    for (int t = 0; t < NT; ++t)
#pragma unroll
        for (int r = 0; r < 8; ++r) acc[t][r] = 0.0f;

    for (int ck = 0; ck < nCk; ++ck) {
        // ---- stage input chunk into LDS (zero halo) ----
        __syncthreads();
        if (Xf) {   // f32 image, layout [b][c][p]: scalar gather + convert
            for (int e = tid; e < 256 * CHUNK; e += 256) {
                const int ci  = e & (CHUNK - 1);
                const int pix = e >> 7;
                const int y = pix >> 4, x = pix & 15;
                unsigned short val = 0;
                if (y >= 1 && y <= 14 && x >= 1 && x <= 14) {
                    const int p  = (y - 1) * 14 + (x - 1);
                    const int cg = ck * CHUNK + ci;
                    val = f2bf_bits(Xf[((size_t)b * CinData + cg) * HW + p]);
                }
                lds[e] = val;
            }
        } else {    // bf16 activations, [b][p][c]: async b128 global->LDS DMA
            for (int v = tid; v < 256 * (CHUNK / 8); v += 256) {
                const int pix = v >> 4;
                const int g   = v & 15;
                const int y = pix >> 4, x = pix & 15;
                if (y >= 1 && y <= 14 && x >= 1 && x <= 14) {
                    const int p  = (y - 1) * 14 + (x - 1);
                    int cg = ck * CHUNK + g * 8;
                    if (cinMod) cg %= cinMod;
                    const unsigned short* src =
                        Xh + ((size_t)b * HW + p) * CinData + cg;
                    const unsigned dst =
                        (unsigned)(uintptr_t)(&lds[pix * CHUNK + g * 8]);
                    async_copy_b128(dst, (unsigned long long)(uintptr_t)src);
                } else {
                    // halo: plain zero store (disjoint from async destinations)
                    *(uint4*)(&lds[pix * CHUNK + g * 8]) = uint4{0u, 0u, 0u, 0u};
                }
            }
            wait_async0();
        }
        __syncthreads();

        for (int tap = 0; tap < KKK; ++tap) {
            const int dh = tap / KK - KK / 2;
            const int dw = tap % KK - KK / 2;
            for (int ks = 0; ks < CHUNK / 32; ++ks) {
                // ---- A fragment: two b128 loads from packed weights ----
                const size_t frag =
                    (((size_t)widx * nCo + cb) * KKK + tap) * nKsTot +
                    (ck * (CHUNK / 32) + ks);
                const unsigned short* ap = Wpk + (frag * 32 + lane) * 16;
                BFrag a;
                a.q[0] = *(const uint4*)(ap);
                a.q[1] = *(const uint4*)(ap + 8);
                __builtin_prefetch(ap + 512, 0, 1);   // next fragment
                // ---- B fragments from LDS (2 x ds_read_b128) + WMMA ----
#pragma unroll
                for (int t = 0; t < NT; ++t) {
                    BFrag bf;
                    const int p = t * 16 + l15;
                    if (p < HW) {
                        const int y = p / 14 + 1 + dh;
                        const int x = p % 14 + 1 + dw;
                        const unsigned short* lp =
                            &lds[(y * 16 + x) * CHUNK + ks * 32 + 8 * half];
                        bf.q[0] = *(const uint4*)(lp);
                        bf.q[1] = *(const uint4*)(lp + 16);
                    } else {
                        bf.q[0] = uint4{0u, 0u, 0u, 0u};
                        bf.q[1] = uint4{0u, 0u, 0u, 0u};
                    }
                    acc[t] = __builtin_amdgcn_wmma_f32_16x16x32_bf16(
                        false, a.v, false, bf.v, (short)0, acc[t], false, false);
                }
            }
        }
    }

    // ---- epilogue: bias (+residual) + relu, one b128 store per tile ----
    float bias8[8];
#pragma unroll
    for (int r = 0; r < 8; ++r) bias8[r] = Bp[rowb + r];

#pragma unroll
    for (int t = 0; t < NT; ++t) {
        const int p = t * 16 + l15;
        if (p >= HW) continue;
        const size_t base = ((size_t)b * HW + p) * Cout + rowb;
        float vals[8];
#pragma unroll
        for (int r = 0; r < 8; ++r) vals[r] = acc[t][r] + bias8[r];
        if (Res) {
            const uint4 rv = *(const uint4*)(Res + base);
            const unsigned int rw[4] = {rv.x, rv.y, rv.z, rv.w};
#pragma unroll
            for (int r = 0; r < 8; ++r) {
                const unsigned short h = (r & 1)
                    ? (unsigned short)(rw[r >> 1] >> 16)
                    : (unsigned short)(rw[r >> 1] & 0xFFFFu);
                vals[r] += bfbits2f(h);
            }
        }
        unsigned int ow[4];
#pragma unroll
        for (int w = 0; w < 4; ++w) {
            const float a0 = vals[2 * w]     < 0.0f ? 0.0f : vals[2 * w];
            const float a1 = vals[2 * w + 1] < 0.0f ? 0.0f : vals[2 * w + 1];
            ow[w] = (unsigned)f2bf_bits(a0) | ((unsigned)f2bf_bits(a1) << 16);
        }
        *(uint4*)(Out + base) = uint4{ow[0], ow[1], ow[2], ow[3]};
    }
}

// ---------------------------------------------------------------------------
// FC1 as WMMA GEMM: out[b][j] = relu( sum_k W[j][k] * P[b][k] + bias[j] )
//   M=1024, K=25088, N=32 (two 16-wide N tiles per wave)
// ---------------------------------------------------------------------------
__global__ __launch_bounds__(256)
void fc1_wmma(const unsigned short* __restrict__ P,  // bf16 [32][K]
              const float* __restrict__ W,           // f32  [M][K]
              const float* __restrict__ Bias,        // [M]
              float* __restrict__ Out,               // f32 [32][M]
              int M, int K)
{
    const int tid  = threadIdx.x;
    const int lane = tid & 31;
    const int wave = tid >> 5;
    const int half = lane >> 4;
    const int l15  = lane & 15;
    const int j_base = (blockIdx.x * 8 + wave) * 16;
    if (j_base >= M) return;

    const int kmapA[8] = {0, 2, 4, 6, 16, 18, 20, 22};

    v8f acc[2];
#pragma unroll
    for (int t = 0; t < 2; ++t)
#pragma unroll
        for (int r = 0; r < 8; ++r) acc[t][r] = 0.0f;

    for (int k0 = 0; k0 < K; k0 += 32) {
        BFrag a;
        const float* wr = W + (size_t)(j_base + l15) * K + k0;
        __builtin_prefetch(wr + 32, 0, 1);
#pragma unroll
        for (int v = 0; v < 8; ++v) {
            const float2 f = *(const float2*)(wr + kmapA[v] + 8 * half);
            a.u16[2 * v]     = f2bf_bits(f.x);
            a.u16[2 * v + 1] = f2bf_bits(f.y);
        }
#pragma unroll
        for (int t = 0; t < 2; ++t) {
            BFrag bf;
            const unsigned short* pp =
                P + (size_t)(t * 16 + l15) * K + k0 + 8 * half;
            bf.q[0] = *(const uint4*)(pp);
            bf.q[1] = *(const uint4*)(pp + 16);
            acc[t] = __builtin_amdgcn_wmma_f32_16x16x32_bf16(
                false, a.v, false, bf.v, (short)0, acc[t], false, false);
        }
    }
#pragma unroll
    for (int t = 0; t < 2; ++t)
#pragma unroll
        for (int r = 0; r < 8; ++r) {
            const int j    = j_base + r + 8 * half;
            const int bcol = t * 16 + l15;
            float v = acc[t][r] + Bias[j];
            if (v < 0.0f) v = 0.0f;
            Out[(size_t)bcol * M + j] = v;
        }
}

// ---------------------------------------------------------------------------
// 2x2 maxpool: clsout bf16 [b][p=196][512] -> flattened bf16 [b][c*49+q]
// ---------------------------------------------------------------------------
__global__ void pool_kernel(const unsigned short* __restrict__ Xin,
                            unsigned short* __restrict__ P, int total)
{
    const int i = blockIdx.x * blockDim.x + threadIdx.x;
    if (i >= total) return;
    const int q = i % 49;
    const int c = (i / 49) % 512;
    const int b = i / (49 * 512);
    const int py = q / 7, px = q % 7;
    const unsigned short* base =
        Xin + ((size_t)b * HW + (2 * py) * 14 + 2 * px) * 512 + c;
    float m = bfbits2f(base[0]);
    m = fmaxf(m, bfbits2f(base[512]));        // x+1
    m = fmaxf(m, bfbits2f(base[14 * 512]));   // y+1
    m = fmaxf(m, bfbits2f(base[15 * 512]));   // y+1, x+1
    P[(size_t)b * 25088 + c * 49 + q] = f2bf_bits(m);
}

// ---------------------------------------------------------------------------
// FC2 (28x1024, tiny) plain VALU
// ---------------------------------------------------------------------------
__global__ void fc2_kernel(const float* __restrict__ X,   // [32][1024]
                           const float* __restrict__ W,   // [28][1024]
                           const float* __restrict__ Bias,
                           float* __restrict__ Out, int total)
{
    const int i = blockIdx.x * blockDim.x + threadIdx.x;
    if (i >= total) return;
    const int c = i % 28, b = i / 28;
    const float* x = X + (size_t)b * 1024;
    const float* w = W + (size_t)c * 1024;
    float s = Bias[c];
    for (int k = 0; k < 1024; ++k) s += x[k] * w[k];
    Out[(size_t)b * 28 + c] = s;
}

// ---------------------------------------------------------------------------
extern "C" void kernel_launch(void* const* d_in, const int* in_sizes, int n_in,
                              void* d_out, int out_size, void* d_ws, size_t ws_size,
                              hipStream_t stream)
{
    const int*   pinds  = (const int*)  d_in[0];
    const float* img    = (const float*)d_in[1];
    const float* cnn_w1 = (const float*)d_in[2];
    const float* cnn_b1 = (const float*)d_in[3];
    const float* cnn_w2 = (const float*)d_in[4];
    const float* cnn_b2 = (const float*)d_in[5];
    const float* un_w1  = (const float*)d_in[6];
    const float* un_b1  = (const float*)d_in[7];
    const float* un_w2  = (const float*)d_in[8];
    const float* un_b2  = (const float*)d_in[9];
    const float* bi_w1  = (const float*)d_in[10];
    const float* bi_b1  = (const float*)d_in[11];
    const float* bi_w2  = (const float*)d_in[12];
    const float* bi_b2  = (const float*)d_in[13];
    const float* bi_w3  = (const float*)d_in[14];
    const float* bi_b3  = (const float*)d_in[15];
    const float* cls_w1 = (const float*)d_in[16];
    const float* cls_b1 = (const float*)d_in[17];
    const float* fc1_w  = (const float*)d_in[18];
    const float* fc1_b  = (const float*)d_in[19];
    const float* fc2_w  = (const float*)d_in[20];
    const float* fc2_b  = (const float*)d_in[21];

    const int B = 32, H = 128;
    const size_t ACT  = (size_t)B * H * HW * 2;       // bf16 activation buffer
    const size_t CLS  = (size_t)B * 512 * HW * 2;
    const size_t POOL = (size_t)B * 25088 * 2;
    const size_t FC1O = (size_t)B * 1024 * 4;

    char* ws = (char*)d_ws;
    size_t off = 0;
    auto alloc = [&](size_t bytes) -> void* {
        void* p = ws + off;
        off += (bytes + 255) & ~(size_t)255;
        return p;
    };
    unsigned short* actA   = (unsigned short*)alloc(ACT);
    unsigned short* actB   = (unsigned short*)alloc(ACT);
    unsigned short* actC   = (unsigned short*)alloc(ACT);
    unsigned short* actD   = (unsigned short*)alloc(ACT);
    unsigned short* clsout = (unsigned short*)alloc(CLS);
    unsigned short* pooled = (unsigned short*)alloc(POOL);
    float*          fc1out = (float*)alloc(FC1O);
    // packed bf16 weights (same element counts as the f32 originals)
    unsigned short* pk_c1  = (unsigned short*)alloc((size_t)128 * 1024 * 9 * 2);
    unsigned short* pk_c2  = (unsigned short*)alloc((size_t)128 * 128 * 9 * 2);
    unsigned short* pk_u1  = (unsigned short*)alloc((size_t)8 * 128 * 128 * 9 * 2);
    unsigned short* pk_u2  = (unsigned short*)alloc((size_t)8 * 128 * 128 * 9 * 2);
    unsigned short* pk_b1  = (unsigned short*)alloc((size_t)4 * 128 * 256 * 2);
    unsigned short* pk_b2  = (unsigned short*)alloc((size_t)4 * 128 * 128 * 9 * 2);
    unsigned short* pk_b3  = (unsigned short*)alloc((size_t)4 * 128 * 128 * 9 * 2);
    unsigned short* pk_cl  = (unsigned short*)alloc((size_t)512 * 128 * 2);
    (void)ws_size; (void)in_sizes; (void)n_in; (void)out_size;

    dim3 blk(256);

    auto pack = [&](const float* W, unsigned short* Wp, int rows, int Cin, int KKK) {
        const int total = (rows / 16) * KKK * (Cin / 32) * 32;
        pack_w_kernel<<<(total + 255) / 256, blk, 0, stream>>>(W, Wp, Cin, KKK, total);
    };
    pack(cnn_w1, pk_c1, 128,     1024, 9);
    pack(cnn_w2, pk_c2, 128,     128,  9);
    pack(un_w1,  pk_u1, 8 * 128, 128,  9);
    pack(un_w2,  pk_u2, 8 * 128, 128,  9);
    pack(bi_w1,  pk_b1, 4 * 128, 256,  1);
    pack(bi_w2,  pk_b2, 4 * 128, 128,  9);
    pack(bi_w3,  pk_b3, 4 * 128, 128,  9);
    pack(cls_w1, pk_cl, 512,     128,  1);

    // CNN stem: conv3x3 1024->128 relu ; conv3x3 128->128 relu
    conv_wmma<<<dim3(B, 1), blk, 0, stream>>>(img, nullptr, pk_c1, cnn_b1,
        nullptr, actA, nullptr, -1, 0, 1024, 1024, 0, 128, 3);
    conv_wmma<<<dim3(B, 1), blk, 0, stream>>>(nullptr, actA, pk_c2, cnn_b2,
        nullptr, actB, nullptr, -1, 0, 128, 128, 0, 128, 3);

    // Binary cell (routed, widx = pinds[:,2]-10): concat handled via cinMod=128
    conv_wmma<<<dim3(B, 1), blk, 0, stream>>>(nullptr, actB, pk_b1, bi_b1,
        nullptr, actA, pinds, 2, 10, 256, 128, 128, 128, 1);
    conv_wmma<<<dim3(B, 1), blk, 0, stream>>>(nullptr, actA, pk_b2, bi_b2,
        nullptr, actC, pinds, 2, 10, 128, 128, 0, 128, 3);
    conv_wmma<<<dim3(B, 1), blk, 0, stream>>>(nullptr, actC, pk_b3, bi_b3,
        actA, actD, pinds, 2, 10, 128, 128, 0, 128, 3);

    // Unary chain, steps 3..7 (routed, widx = pinds[:,s]-2)
    unsigned short* cur = actD;
    unsigned short* hbf = actA;
    unsigned short* nxt = actB;   // feats no longer needed after bi_w1 conv
    for (int s = 3; s < 8; ++s) {
        conv_wmma<<<dim3(B, 1), blk, 0, stream>>>(nullptr, cur, pk_u1, un_b1,
            nullptr, hbf, pinds, s, 2, 128, 128, 0, 128, 3);
        conv_wmma<<<dim3(B, 1), blk, 0, stream>>>(nullptr, hbf, pk_u2, un_b2,
            cur, nxt, pinds, s, 2, 128, 128, 0, 128, 3);
        unsigned short* t = cur; cur = nxt; nxt = t;
    }

    // Classifier: conv1x1 128->512 relu
    conv_wmma<<<dim3(B, 4), blk, 0, stream>>>(nullptr, cur, pk_cl, cls_b1,
        nullptr, clsout, nullptr, -1, 0, 128, 128, 0, 512, 1);

    // 2x2 maxpool + flatten
    {
        const int total = B * 512 * 49;
        pool_kernel<<<(total + 255) / 256, blk, 0, stream>>>(clsout, pooled, total);
    }

    // FC1 (WMMA) + relu
    fc1_wmma<<<8, blk, 0, stream>>>(pooled, fc1_w, fc1_b, fc1out, 1024, 25088);

    // FC2 -> d_out  [32][28] f32
    {
        const int total = B * 28;
        fc2_kernel<<<(total + 255) / 256, blk, 0, stream>>>(
            fc1out, fc2_w, fc2_b, (float*)d_out, total);
    }
}
